// Classification_GCN_84739704750845
// MI455X (gfx1250) — matrix operand order repaired and checked by reading, not compiled, not run on hardware
//
#include <hip/hip_runtime.h>

typedef _Float16 v16h __attribute__((ext_vector_type(16)));
typedef _Float16 v8h  __attribute__((ext_vector_type(8)));
typedef float    v8f  __attribute__((ext_vector_type(8)));

struct U32x8 { uint4 a, b; };
__device__ __forceinline__ v16h join16(uint4 a, uint4 b) {
  U32x8 t{a, b};
  return __builtin_bit_cast(v16h, t);
}

struct __align__(8) H4 { _Float16 v[4]; };

#define SAMP  16
#define WAVES 2
#define SS    72              // sample stride (halves): 144B, 16B-aligned, bank-spread
#define NS    (SAMP * SS)     // node stride = 1152 halves
#define HBUF  (6 * NS)        // 6912 halves per wave buffer

// One GCN layer for one wave's 16-sample tile.
// src/dst: f16 LDS activations, layout [node][sample(SS)][feat].
// WMMA roles: A = weight tile (M = out-feature), B = aggregated activations
// (N = sample)  ->  D lane holds 8 consecutive out-features of one sample.
// Epilogue fully packed-f16: cvt_pk -> pk_add(bias) -> pk_max(relu) -> b128.
template<int IMP>
__device__ __forceinline__ void layer_step(const _Float16* __restrict__ src,
                                           _Float16* __restrict__ dst,
                                           const _Float16* __restrict__ Wh,
                                           const float* __restrict__ bias,
                                           int srow, int hi) {
  // ---- load activation B-fragments for the 6 source nodes -----------------
  // dense 16-bit B 32x16: lane holds column (sample)=srow, K = hi*16..+15
  v16h fb0[6], fb1[6];
#pragma unroll
  for (int m = 0; m < 6; ++m) {
    const _Float16* p = src + m * NS + srow * SS + hi * 16;
    fb0[m] = join16(*(const uint4*)(p),      *(const uint4*)(p + 8));   // K 0..31
    fb1[m] = join16(*(const uint4*)(p + 32), *(const uint4*)(p + 40));  // K 32..63
  }

  // ---- aggregate: per destination node, linear combo of source fragments --
  v16h ra0[6], ra1[6];
  if (IMP == 0) {
    const _Float16 c3 = (_Float16)(1.0f / 3.0f);
    v16h s12a = fb0[1] + fb0[2], s12b = fb1[1] + fb1[2];
    v16h s34a = fb0[3] + fb0[4], s34b = fb1[3] + fb1[4];
    ra0[0] = (fb0[0] + s12a) * c3;  ra1[0] = (fb1[0] + s12b) * c3;
    ra0[1] = ra0[0];                ra1[1] = ra1[0];
    ra0[2] = (s12a + fb0[3]) * c3;  ra1[2] = (s12b + fb1[3]) * c3;
    ra0[3] = (fb0[2] + s34a) * c3;  ra1[3] = (fb1[2] + s34b) * c3;
    ra0[4] = (s34a + fb0[5]) * c3;  ra1[4] = (s34b + fb1[5]) * c3;
    ra0[5] = ra0[4];                ra1[5] = ra1[4];
  } else {
    const _Float16 c2 = (_Float16)0.5f, c4 = (_Float16)0.25f;
    // neighbor pairs for rows 0..5 of the improved adjacency
    constexpr int nb[6][2] = {{1,2},{0,2},{1,3},{2,4},{3,5},{3,4}};
#pragma unroll
    for (int n = 0; n < 6; ++n) {
      ra0[n] = (fb0[nb[n][0]] + fb0[nb[n][1]]) * c4 + fb0[n] * c2;
      ra1[n] = (fb1[nb[n][0]] + fb1[nb[n][1]]) * c4 + fb1[n] * c2;
    }
  }

  // ---- weight A-fragments (loop-invariant over n) --------------------------
  // 16-bit A 16x32: lane holds row j = t*16+srow, K = hi*16..+15
  v16h wa0[4], wa1[4];
#pragma unroll
  for (int t = 0; t < 4; ++t) {
    const int j = t * 16 + srow;
    const _Float16* p = Wh + j * 64 + hi * 16;
    wa0[t] = join16(*(const uint4*)(p),      *(const uint4*)(p + 8));
    wa1[t] = join16(*(const uint4*)(p + 32), *(const uint4*)(p + 40));
  }

  // ---- bias pre-converted to packed f16, hoisted out of the n-loop ---------
  v8h bias16[4];
#pragma unroll
  for (int t = 0; t < 4; ++t) {
    const float4 bA = *(const float4*)(bias + t * 16 + hi * 8);
    const float4 bB = *(const float4*)(bias + t * 16 + hi * 8 + 4);
    v8h b;
    b[0] = (_Float16)bA.x; b[1] = (_Float16)bA.y;
    b[2] = (_Float16)bA.z; b[3] = (_Float16)bA.w;
    b[4] = (_Float16)bB.x; b[5] = (_Float16)bB.y;
    b[6] = (_Float16)bB.z; b[7] = (_Float16)bB.w;
    bias16[t] = b;
  }
  v8h hz;
#pragma unroll
  for (int r = 0; r < 8; ++r) hz[r] = (_Float16)0.0f;
  v8f zc;
#pragma unroll
  for (int r = 0; r < 8; ++r) zc[r] = 0.0f;

#pragma unroll
  for (int n = 0; n < 6; ++n) {
#pragma unroll
    for (int t = 0; t < 4; ++t) {
      v8f acc = __builtin_amdgcn_wmma_f32_16x16x32_f16(false, wa0[t], false, ra0[n],
                                                       (short)0, zc, false, false);
      acc = __builtin_amdgcn_wmma_f32_16x16x32_f16(false, wa1[t], false, ra1[n],
                                                   (short)0, acc, false, false);
      // D: lane = sample srow, M = t*16 + hi*8 + r  -> contiguous features
      v8h hv;
#pragma unroll
      for (int r = 0; r < 8; ++r) hv[r] = (_Float16)acc[r];   // 4x cvt_pk
      hv = hv + bias16[t];                                    // 4x pk_add
      hv = __builtin_elementwise_max(hv, hz);                 // 4x pk_max (relu)
      *(v8h*)(dst + n * NS + srow * SS + t * 16 + hi * 8) = hv;
    }
  }
}

__global__ void convw_f16(const float* __restrict__ W1, const float* __restrict__ W2,
                          const float* __restrict__ W3, const float* __restrict__ W4,
                          _Float16* __restrict__ o) {
  const int i = blockIdx.x * 256 + threadIdx.x;      // 0..16383
  const float* Ws[4] = {W1, W2, W3, W4};
  o[i] = (_Float16)Ws[i >> 12][i & 4095];
}

__global__ __launch_bounds__(64) void gcn6(const float* __restrict__ x,
    const float* __restrict__ b1, const float* __restrict__ b2,
    const float* __restrict__ b3, const float* __restrict__ b4,
    const float* __restrict__ fcw, const float* __restrict__ fcb,
    const _Float16* __restrict__ Wh, float* __restrict__ out) {
  __shared__ _Float16 hbuf[2][WAVES * HBUF];   // ping-pong activations (f16)
  __shared__ float    px[WAVES][SAMP][6];      // residual partial dots

  const int tid  = threadIdx.x;
  const int wave = tid >> 5, lane = tid & 31;
  const int srow = lane & 15, hi = lane >> 4;
  const long sbase = (long)(blockIdx.x * WAVES + wave) * SAMP;

  _Float16* buf0 = &hbuf[0][wave * HBUF];
  _Float16* buf1 = &hbuf[1][wave * HBUF];

  float* pxf = &px[wave][0][0];
  for (int i = lane; i < SAMP * 6; i += 32) pxf[i] = 0.0f;
  __syncthreads();

  // Load x -> buf0 (f16) and fold residual head: px[s][n] += dot(x, fc_w[n])
#pragma unroll 4
  for (int i = 0; i < 48; ++i) {
    const int c  = lane + i * 32;
    const int kc = c & 15;
    const int sn = c >> 4;
    const int s = sn & 15, n = sn >> 4;
    const float4 xv = *(const float4*)(x + ((sbase + s) * 6 + n) * 64 + kc * 4);
    const float4 fw = *(const float4*)(fcw + n * 64 + kc * 4);
    atomicAdd(&px[wave][s][n],
              xv.x * fw.x + xv.y * fw.y + xv.z * fw.z + xv.w * fw.w);
    H4 h;
    h.v[0] = (_Float16)xv.x; h.v[1] = (_Float16)xv.y;
    h.v[2] = (_Float16)xv.z; h.v[3] = (_Float16)xv.w;
    *(H4*)(buf0 + n * NS + s * SS + kc * 4) = h;
  }
  __syncthreads();

  layer_step<0>(buf0, buf1, Wh,         b1, srow, hi); __syncthreads();
  layer_step<0>(buf1, buf0, Wh + 4096,  b2, srow, hi); __syncthreads();
  layer_step<1>(buf0, buf1, Wh + 8192,  b3, srow, hi); __syncthreads();
  layer_step<1>(buf1, buf0, Wh + 12288, b4, srow, hi); __syncthreads();
  // final activations in buf0

  // Head: logits[s][n] = dot(h4[s][n], fc_w[n]) + px[s][n] + fc_b[n]
  float lg[6];
#pragma unroll
  for (int n = 0; n < 6; ++n) {
    float acc = 0.0f;
#pragma unroll
    for (int q = 0; q < 4; ++q) {
      const v8h hv = *(const v8h*)(buf0 + n * NS + srow * SS + hi * 32 + q * 8);
      const float4 f0 = *(const float4*)(fcw + n * 64 + hi * 32 + q * 8);
      const float4 f1 = *(const float4*)(fcw + n * 64 + hi * 32 + q * 8 + 4);
      acc += (float)hv[0] * f0.x + (float)hv[1] * f0.y +
             (float)hv[2] * f0.z + (float)hv[3] * f0.w +
             (float)hv[4] * f1.x + (float)hv[5] * f1.y +
             (float)hv[6] * f1.z + (float)hv[7] * f1.w;
    }
    acc += __shfl_xor(acc, 16);
    lg[n] = acc + px[wave][srow][n] + fcb[n];
  }
  if (hi == 0) {
    float o[6];
#pragma unroll
    for (int n = 0; n < 6; ++n) o[n] = 1.0f / (1.0f + __expf(-lg[n]));
    float* po = out + (sbase + srow) * 6;
    *(float2*)(po + 0) = make_float2(o[0], o[1]);
    *(float2*)(po + 2) = make_float2(o[2], o[3]);
    *(float2*)(po + 4) = make_float2(o[4], o[5]);
  }
}

extern "C" void kernel_launch(void* const* d_in, const int* in_sizes, int n_in,
                              void* d_out, int out_size, void* d_ws, size_t ws_size,
                              hipStream_t stream) {
  const float* x   = (const float*)d_in[0];
  const float* W1  = (const float*)d_in[1];
  const float* b1  = (const float*)d_in[2];
  const float* W2  = (const float*)d_in[3];
  const float* b2  = (const float*)d_in[4];
  const float* W3  = (const float*)d_in[5];
  const float* b3  = (const float*)d_in[6];
  const float* W4  = (const float*)d_in[7];
  const float* b4  = (const float*)d_in[8];
  const float* fcw = (const float*)d_in[9];
  const float* fcb = (const float*)d_in[10];
  _Float16* Wh = (_Float16*)d_ws;          // 4*64*64 halves = 32 KB
  float* out = (float*)d_out;

  hipLaunchKernelGGL(convw_f16, dim3(64), dim3(256), 0, stream, W1, W2, W3, W4, Wh);
  // 131072 samples / (2 waves * 16 samples) = 4096 blocks of 64 threads
  hipLaunchKernelGGL(gcn6, dim3(4096), dim3(64), 0, stream,
                     x, b1, b2, b3, b4, fcw, fcb, Wh, out);
}
